// CausalSelfAttention_29274497090296
// MI455X (gfx1250) — compile-verified, matrix operand order checked by reading, not run on hardware
//
#include <hip/hip_runtime.h>
#include <hip/hip_bf16.h>

// ---------------------------------------------------------------------------
// CausalSelfAttention for MI455X (gfx1250): bf16 WMMA + TDM async tile loads.
// T=2048, B=2, C=1024, H=16, D=64.
// ---------------------------------------------------------------------------

#define T_SEQ 2048
#define BATCH 2
#define DMODEL 1024
#define NHEADS 16
#define HDIM 64
#define MTOT (T_SEQ * BATCH)   // 4096 rows in (T,B,C) flattening

typedef __attribute__((ext_vector_type(16))) __bf16 bf16x16;
typedef __attribute__((ext_vector_type(8)))  __bf16 bf16x8;
typedef __attribute__((ext_vector_type(8)))  float  f32x8;
typedef __attribute__((ext_vector_type(4)))  unsigned int u32x4;
typedef __attribute__((ext_vector_type(4)))  int  i32x4;
typedef __attribute__((ext_vector_type(8)))  int  i32x8;

#if defined(__HIP_DEVICE_COMPILE__) && __has_builtin(__builtin_amdgcn_tensor_load_to_lds) && __has_builtin(__builtin_amdgcn_s_wait_tensorcnt)
#define HAVE_TDM 1
#else
#define HAVE_TDM 0
#endif

// ---- float -> bf16 (round to nearest even), pure integer ------------------
__device__ __forceinline__ unsigned short f2bf(float f) {
    union { float f; unsigned u; } v; v.f = f;
    unsigned u = v.u;
    u = u + 0x7FFFu + ((u >> 16) & 1u);
    return (unsigned short)(u >> 16);
}

// ---- WMMA bf16 wrapper ----------------------------------------------------
__device__ __forceinline__ f32x8 wmma_bf16(bf16x16 a, bf16x16 b, f32x8 c) {
    return __builtin_amdgcn_wmma_f32_16x16x32_bf16(
        /*neg_a=*/false, a, /*neg_b=*/false, b,
        /*c_mod=*/(short)0, c, /*reuse_a=*/false, /*reuse_b=*/false);
}

// ---- Load a 16x32 bf16 operand fragment (A layout; B mirrored on N) -------
// lanes 0-15 = row lane, K {0..7,16..23}; lanes 16-31 = row lane-16, complement.
__device__ __forceinline__ bf16x16 load_frag(const unsigned short* p, int ld) {
    int lane = threadIdx.x & 31;
    const __bf16* row = (const __bf16*)(p + (size_t)(lane & 15) * ld) + ((lane >> 4) * 8);
    bf16x8 lo = *(const bf16x8*)(row);
    bf16x8 hi = *(const bf16x8*)(row + 16);
    return __builtin_shufflevector(lo, hi, 0,1,2,3,4,5,6,7,8,9,10,11,12,13,14,15);
}

#if HAVE_TDM
// ---- TDM: 2D tile (tile0 elems/row x tile1 rows, 2B elems) global -> LDS --
// D# per CDNA5 ISA. Tensor dims == tile dims so the
// OOB clamp never triggers; row stride in elements; tile rows land compact.
__device__ __forceinline__ void tdm_load_2d(const void* gsrc, unsigned lds_byte,
                                            unsigned tile0, unsigned tile1,
                                            unsigned long long stride0) {
    unsigned long long ga = (unsigned long long)(size_t)gsrc;
    u32x4 g0 = {};
    g0[0] = 1u;                                   // count=1, user descriptor
    g0[1] = lds_byte;                             // lds_addr (bytes)
    g0[2] = (unsigned)ga;                         // global_addr[31:0]
    g0[3] = (unsigned)((ga >> 32) & 0x1FFFFFFu) | 0x80000000u;  // addr[56:32], type=2
    unsigned w[8] = {0,0,0,0,0,0,0,0};
    w[0] = 1u << 16;                              // data_size = 1 (2 bytes)
    w[1] = (tile0 & 0xFFFFu) << 16;               // tensor_dim0[15:0]   (bits 79:48)
    w[2] = (tile0 >> 16) | ((tile1 & 0xFFFFu) << 16);   // td0 hi | tensor_dim1 lo
    w[3] = (tile1 >> 16) | ((tile0 & 0xFFFFu) << 16);   // td1 hi | tile_dim0
    w[4] = (tile1 & 0xFFFFu);                     // tile_dim1 (tile_dim2 = 0)
    w[5] = (unsigned)(stride0 & 0xFFFFFFFFull);   // tensor_dim0_stride[31:0]
    w[6] = (unsigned)((stride0 >> 32) & 0xFFFFu); // stride hi (dim1_stride = 0)
    w[7] = 0u;
    i32x8 g1;
    #pragma unroll
    for (int i = 0; i < 8; ++i) g1[i] = (int)w[i];
    i32x4 z4 = {};
    i32x8 z8 = {};
    (void)z8;
#if defined(__clang_major__) && (__clang_major__ >= 23)
    __builtin_amdgcn_tensor_load_to_lds(g0, g1, z4, z4, z8, 0);
#else
    __builtin_amdgcn_tensor_load_to_lds(g0, g1, z4, z4, 0);
#endif
}
#endif

// ---------------------------------------------------------------------------
// Cast fp32 -> bf16, 4 elements per thread.
// ---------------------------------------------------------------------------
__global__ void cast_bf16_kernel(const float* __restrict__ in,
                                 unsigned short* __restrict__ out, int n4) {
    int i = blockIdx.x * blockDim.x + threadIdx.x;
    if (i >= n4) return;
    float4 f = ((const float4*)in)[i];
    uint2 o;
    o.x = (unsigned)f2bf(f.x) | ((unsigned)f2bf(f.y) << 16);
    o.y = (unsigned)f2bf(f.z) | ((unsigned)f2bf(f.w) << 16);
    *(uint2*)(out + 4 * (size_t)i) = o;
}

// ---------------------------------------------------------------------------
// GEMM: out[m,n] = sum_k A[m,k] * W[n,k]   (A: 4096xK, W: NxK, K=N=1024)
// TDM double-buffered LDS tiles: A 128x64, W 64x64 per k-chunk (48 KB LDS).
// Block 256 thr = 8 waves (4x2), wave tile 32x32 (2x2 WMMA).
// mode 0: bf16 out in per-head [b][h][t][d]; mode 1: fp32 row-major.
// Grid: (32, 16).
// ---------------------------------------------------------------------------
__global__ __launch_bounds__(256) void gemm_bf16_kernel(
        const unsigned short* __restrict__ A,
        const unsigned short* __restrict__ W,
        unsigned short* __restrict__ out_qkv,
        float* __restrict__ out_f32,
        int mode) {
    __shared__ unsigned short Abuf[2][128 * 64];   // [buf][row][k]  32 KB
    __shared__ unsigned short Wbuf[2][64 * 64];    // [buf][row][k]  16 KB

    const int K = DMODEL;
    const int NCH = K / 64;                        // 16 k-chunks
    int wid  = threadIdx.x >> 5;
    int wm   = wid & 3;                            // 4 wave rows
    int wn   = wid >> 2;                           // 2 wave cols
    int mblk = blockIdx.x * 128;
    int nblk = blockIdx.y * 64;
    int m0   = wm * 32;                            // local tile coords
    int n0   = wn * 32;
    int lane = threadIdx.x & 31;
    int half = lane >> 4, col = lane & 15;

    f32x8 acc[2][2] = {};

    // ---- chunk loader: global -> LDS buf ---------------------------------
    auto load_chunk = [&](int c) {
        int kc = c * 64;
        int buf = c & 1;
#if HAVE_TDM
        if (threadIdx.x < 32) {
            tdm_load_2d(A + (size_t)mblk * K + kc,
                        (unsigned)(size_t)&Abuf[buf][0], 64u, 128u, (unsigned long long)K);
            tdm_load_2d(W + (size_t)nblk * K + kc,
                        (unsigned)(size_t)&Wbuf[buf][0], 64u, 64u, (unsigned long long)K);
        }
#else
        // cooperative fallback: 16B per chunk, A = 1024 chunks, W = 512 chunks
        for (int i = threadIdx.x; i < 1024; i += 256) {
            int r = i >> 3, co = (i & 7) * 8;
            *(uint4*)&Abuf[buf][r * 64 + co] =
                *(const uint4*)(A + (size_t)(mblk + r) * K + kc + co);
        }
        for (int i = threadIdx.x; i < 512; i += 256) {
            int r = i >> 3, co = (i & 7) * 8;
            *(uint4*)&Wbuf[buf][r * 64 + co] =
                *(const uint4*)(W + (size_t)(nblk + r) * K + kc + co);
        }
#endif
    };

    load_chunk(0);
    for (int c = 0; c < NCH; ++c) {
        if (c + 1 < NCH) load_chunk(c + 1);        // overlap TDM with compute
#if HAVE_TDM
        if (threadIdx.x < 32) {
            if (c + 1 < NCH) __builtin_amdgcn_s_wait_tensorcnt(2);  // chunk c done
            else             __builtin_amdgcn_s_wait_tensorcnt(0);
        }
#endif
        __syncthreads();
        const unsigned short* At = Abuf[c & 1];
        const unsigned short* Wt = Wbuf[c & 1];
        #pragma unroll
        for (int ks = 0; ks < 64; ks += 32) {
            bf16x16 a0 = load_frag(At + (size_t)(m0)      * 64 + ks, 64);
            bf16x16 a1 = load_frag(At + (size_t)(m0 + 16) * 64 + ks, 64);
            bf16x16 b0 = load_frag(Wt + (size_t)(n0)      * 64 + ks, 64);
            bf16x16 b1 = load_frag(Wt + (size_t)(n0 + 16) * 64 + ks, 64);
            acc[0][0] = wmma_bf16(a0, b0, acc[0][0]);
            acc[0][1] = wmma_bf16(a0, b1, acc[0][1]);
            acc[1][0] = wmma_bf16(a1, b0, acc[1][0]);
            acc[1][1] = wmma_bf16(a1, b1, acc[1][1]);
        }
        __syncthreads();
    }

    for (int i = 0; i < 2; ++i)
        for (int j = 0; j < 2; ++j)
            for (int r = 0; r < 8; ++r) {
                int m = mblk + m0 + i * 16 + r + 8 * half;
                int n = nblk + n0 + j * 16 + col;
                float v = acc[i][j][r];
                if (mode == 0) {
                    int t = m >> 1, b = m & 1;     // m = t*B + b, B=2
                    int h = n >> 6, d = n & 63;    // n = h*64 + d
                    out_qkv[(((size_t)(b * NHEADS + h)) * T_SEQ + t) * HDIM + d] = f2bf(v);
                } else {
                    out_f32[(size_t)m * DMODEL + n] = v;
                }
            }
}

// ---------------------------------------------------------------------------
// Flash attention, one (b,h) head, 64 query rows per block (4 waves x 16 q).
// Q/K/V in bf16 layout [b*H+h][T][64]. ctx out bf16 in (T,B,C) layout.
// Grid: (T/64, B*H) = (32, 32), block 128.
// ---------------------------------------------------------------------------
__global__ __launch_bounds__(128) void attn_kernel(
        const unsigned short* __restrict__ Qm,
        const unsigned short* __restrict__ Km,
        const unsigned short* __restrict__ Vm,
        unsigned short* __restrict__ ctx) {
    __shared__ unsigned short Vt[HDIM * 32];          // V^T tile: [d][t_local]
    __shared__ unsigned short Plds[4 * 16 * 32];      // per-wave P staging

    int bh   = blockIdx.y;
    int q0   = blockIdx.x * 64;
    int w    = threadIdx.x >> 5;
    int lane = threadIdx.x & 31;
    int half = lane >> 4, col = lane & 15;
    int q0w  = q0 + w * 16;

    const unsigned short* Qb = Qm + ((size_t)bh * T_SEQ + q0w) * HDIM;
    bf16x16 qf0 = load_frag(Qb,      HDIM);
    bf16x16 qf1 = load_frag(Qb + 32, HDIM);

    f32x8 o[4] = {};
    float m_i[8], l_i[8];
    for (int r = 0; r < 8; ++r) { m_i[r] = -3.0e38f; l_i[r] = 0.0f; }

    const float BIGNEG = -3.0e38f;
    int kend = q0 + 64;                 // exclusive causal key limit for block

    for (int kb = 0; kb < kend; kb += 32) {
        __syncthreads();
        // ---- stage V^T tile in LDS: Vt[d][t] for t in [kb, kb+32) ---------
        for (int c = 0; c < 2; ++c) {
            int chunk = threadIdx.x + c * 128;        // 0..255 = 32 t x 8 dchunk
            int tl = chunk >> 3;
            int dc = (chunk & 7) * 8;
            alignas(16) unsigned short tmp[8];
            *(uint4*)tmp = *(const uint4*)(Vm + ((size_t)bh * T_SEQ + kb + tl) * HDIM + dc);
            #pragma unroll
            for (int i = 0; i < 8; ++i) Vt[(dc + i) * 32 + tl] = tmp[i];
        }
        __syncthreads();

        // ---- S = Q (16x64) @ K^T (64x32), two 16-key column tiles ---------
        f32x8 s[2];
        #pragma unroll
        for (int j = 0; j < 2; ++j) {
            const unsigned short* Kb = Km + ((size_t)bh * T_SEQ + kb + j * 16) * HDIM;
            f32x8 z = {};
            z = wmma_bf16(qf0, load_frag(Kb,      HDIM), z);
            z = wmma_bf16(qf1, load_frag(Kb + 32, HDIM), z);
            s[j] = z;
        }

        // ---- scale + causal mask + per-row max ----------------------------
        float rmax[8];
        #pragma unroll
        for (int r = 0; r < 8; ++r) {
            int qg = q0w + r + 8 * half;
            float a0 = s[0][r] * 0.125f;              // 1/sqrt(64)
            float a1 = s[1][r] * 0.125f;
            if (kb + col      > qg) a0 = BIGNEG;
            if (kb + 16 + col > qg) a1 = BIGNEG;
            s[0][r] = a0; s[1][r] = a1;
            rmax[r] = fmaxf(a0, a1);
        }
        #pragma unroll
        for (int off = 1; off < 16; off <<= 1)
            #pragma unroll
            for (int r = 0; r < 8; ++r)
                rmax[r] = fmaxf(rmax[r], __shfl_xor(rmax[r], off, 32));

        // ---- online softmax update; stash P (bf16) in per-wave LDS --------
        unsigned short* pP = Plds + w * (16 * 32);
        float rsum[8];
        #pragma unroll
        for (int r = 0; r < 8; ++r) {
            float mn = fmaxf(m_i[r], rmax[r]);
            float sc = __expf(m_i[r] - mn);
            m_i[r] = mn;
            float p0 = __expf(s[0][r] - mn);
            float p1 = __expf(s[1][r] - mn);
            rsum[r] = p0 + p1;
            l_i[r] *= sc;
            o[0][r] *= sc; o[1][r] *= sc; o[2][r] *= sc; o[3][r] *= sc;
            int prow = (r + 8 * half) * 32;
            pP[prow + col]      = f2bf(p0);
            pP[prow + 16 + col] = f2bf(p1);
        }
        #pragma unroll
        for (int off = 1; off < 16; off <<= 1)
            #pragma unroll
            for (int r = 0; r < 8; ++r)
                rsum[r] += __shfl_xor(rsum[r], off, 32);
        #pragma unroll
        for (int r = 0; r < 8; ++r) l_i[r] += rsum[r];

        // ---- O += P (16x32) @ V (32x64): 4 d-tiles from LDS ---------------
        bf16x16 pf = load_frag(pP, 32);
        #pragma unroll
        for (int dt = 0; dt < 4; ++dt) {
            bf16x16 vf = load_frag(Vt + (size_t)(dt * 16) * 32, 32);
            o[dt] = wmma_bf16(pf, vf, o[dt]);
        }
    }

    // ---- normalize and write ctx in (T,B,C) bf16 --------------------------
    int b = bh >> 4, h = bh & 15;
    #pragma unroll
    for (int r = 0; r < 8; ++r) {
        float inv = 1.0f / l_i[r];
        int t = q0w + r + 8 * half;
        size_t rowoff = ((size_t)t * BATCH + b) * DMODEL + h * HDIM;
        #pragma unroll
        for (int dt = 0; dt < 4; ++dt)
            ctx[rowoff + dt * 16 + col] = f2bf(o[dt][r] * inv);
    }
}

// ---------------------------------------------------------------------------
extern "C" void kernel_launch(void* const* d_in, const int* in_sizes, int n_in,
                              void* d_out, int out_size, void* d_ws, size_t ws_size,
                              hipStream_t stream) {
    const float* x  = (const float*)d_in[0];   // (T,B,C)
    const float* Wq = (const float*)d_in[1];
    const float* Wk = (const float*)d_in[2];
    const float* Wv = (const float*)d_in[3];
    const float* Wo = (const float*)d_in[4];

    // workspace carve (all bf16 as u16); total = 48 MB
    char* p = (char*)d_ws;
    unsigned short* xbf  = (unsigned short*)p; p += (size_t)MTOT * DMODEL * 2;      // 8 MB
    unsigned short* wqb  = (unsigned short*)p; p += (size_t)DMODEL * DMODEL * 2;    // 2 MB
    unsigned short* wkb  = (unsigned short*)p; p += (size_t)DMODEL * DMODEL * 2;
    unsigned short* wvb  = (unsigned short*)p; p += (size_t)DMODEL * DMODEL * 2;
    unsigned short* wob  = (unsigned short*)p; p += (size_t)DMODEL * DMODEL * 2;
    unsigned short* Qb   = (unsigned short*)p; p += (size_t)BATCH * NHEADS * T_SEQ * HDIM * 2; // 8 MB
    unsigned short* Kb   = (unsigned short*)p; p += (size_t)BATCH * NHEADS * T_SEQ * HDIM * 2;
    unsigned short* Vb   = (unsigned short*)p; p += (size_t)BATCH * NHEADS * T_SEQ * HDIM * 2;
    unsigned short* ctxb = (unsigned short*)p; p += (size_t)MTOT * DMODEL * 2;      // 8 MB

    // casts to bf16
    {
        int n4x = MTOT * DMODEL / 4;            // 1048576
        int n4w = DMODEL * DMODEL / 4;          // 262144
        cast_bf16_kernel<<<(n4x + 255) / 256, 256, 0, stream>>>(x,  xbf, n4x);
        cast_bf16_kernel<<<(n4w + 255) / 256, 256, 0, stream>>>(Wq, wqb, n4w);
        cast_bf16_kernel<<<(n4w + 255) / 256, 256, 0, stream>>>(Wk, wkb, n4w);
        cast_bf16_kernel<<<(n4w + 255) / 256, 256, 0, stream>>>(Wv, wvb, n4w);
        cast_bf16_kernel<<<(n4w + 255) / 256, 256, 0, stream>>>(Wo, wob, n4w);
    }

    // QKV projections -> per-head bf16 layouts
    dim3 gg(MTOT / 128, DMODEL / 64);           // (32,16)
    gemm_bf16_kernel<<<gg, 256, 0, stream>>>(xbf, wqb, Qb, nullptr, 0);
    gemm_bf16_kernel<<<gg, 256, 0, stream>>>(xbf, wkb, Kb, nullptr, 0);
    gemm_bf16_kernel<<<gg, 256, 0, stream>>>(xbf, wvb, Vb, nullptr, 0);

    // causal flash attention
    dim3 ga(T_SEQ / 64, BATCH * NHEADS);        // (32,32)
    attn_kernel<<<ga, 128, 0, stream>>>(Qb, Kb, Vb, ctxb);

    // output projection -> fp32 d_out
    gemm_bf16_kernel<<<gg, 256, 0, stream>>>(ctxb, wob, nullptr, (float*)d_out, 1);
}
